// MagnitudeRasterizer2D_7791070675070
// MI455X (gfx1250) — matrix operand order repaired and checked by reading, not compiled, not run on hardware
//
#include <hip/hip_runtime.h>
#include <hip/hip_bf16.h>

typedef __attribute__((ext_vector_type(16))) _Float16 v16h;
typedef __attribute__((ext_vector_type(8)))  float    v8f;

#define IMG_H 224
#define IMG_W 224
#define MAXN  1024
#define STRIPS_PER_ROW (IMG_W / 16)             // 14
#define TOTAL_STRIPS   (IMG_H * STRIPS_PER_ROW) // 3136
#define WAVES_PER_BLOCK 8

// exp(-0.5*d) = exp2(KS*d),  KS = -0.5*log2(e)
#define KSCALE  (-0.7213475204444817f)
#define QCLAMP  (30.0f * KSCALE)                // = -21.640425613f (min(d,30) -> max(q, 30*KS))

// ---------------------------------------------------------------------------
// Kernel 1: per-gaussian preprocessing.
// Emits 8 floats per gaussian into workspace:
//   [cx, cy, KS*i00, KS*2*i01, KS*i11, a*c0, a*c1, a*c2]
// (exp scale folded into the quadratic-form coefficients)
// ---------------------------------------------------------------------------
__global__ void gauss_prep_kernel(const float* __restrict__ alpha,
                                  const float* __restrict__ color,
                                  const float* __restrict__ offset,
                                  const float* __restrict__ scale,
                                  const float* __restrict__ rotation,
                                  const float* __restrict__ basep,
                                  const float* __restrict__ imp,
                                  float* __restrict__ gp, int n) {
    int g = blockIdx.x * blockDim.x + threadIdx.x;
    if (g >= n) return;
    const float lim = 224.0f;
    float cx = fminf(fmaxf(basep[2*g+0] + offset[2*g+0], 0.0f), lim);
    float cy = fminf(fmaxf(basep[2*g+1] + offset[2*g+1], 0.0f), lim);
    float r  = rotation[g];
    float c  = cosf(r), s = sinf(r);
    float sx = scale[2*g+0], sy = scale[2*g+1];
    float sx2 = sx * sx, sy2 = sy * sy;
    // cov = R diag(sx^2, sy^2) R^T + 1e-4 I
    float a00 = c*c*sx2 + s*s*sy2 + 1e-4f;
    float a11 = s*s*sx2 + c*c*sy2 + 1e-4f;
    float a01 = c*s*(sx2 - sy2);
    float det = a00*a11 - a01*a01;
    float inv = 1.0f / det;
    float amp = alpha[g] * imp[g];
    float* o = gp + (size_t)g * 8;
    o[0] = cx;  o[1] = cy;
    o[2] = KSCALE * ( a11 * inv);            // KS * i00
    o[3] = KSCALE * (-2.0f * a01 * inv);     // KS * 2*i01
    o[4] = KSCALE * ( a00 * inv);            // KS * i11
    o[5] = amp * color[3*g+0];
    o[6] = amp * color[3*g+1];
    o[7] = amp * color[3*g+2];
}

// ---------------------------------------------------------------------------
// Weight tile for one 16x32 f16 A-matrix chunk (VGPR layout per ISA:
// lanes 0-15 -> K = {j, j+8(j>=8)}, lanes 16-31 -> +8).
// q = KS*(d^T Sinv d) computed directly with pre-scaled coefficients;
// weight = exp2(max(q, 30*KS)).
// ---------------------------------------------------------------------------
__device__ __forceinline__ v16h weight_tile(const float* __restrict__ sh,
                                            int kbase, int hioff,
                                            float px, float py) {
    v16h av;
#pragma unroll
    for (int j = 0; j < 16; ++j) {
        int k = kbase + j + ((j >= 8) ? 8 : 0) + hioff;
        const float* p = &sh[k * 8];
        float dx = px - p[0];
        float dy = py - p[1];
        float q = fmaf(p[2] * dx, dx, fmaf(p[3] * dx, dy, p[4] * dy * dy));
        av[j] = (_Float16)__builtin_amdgcn_exp2f(fmaxf(q, QCLAMP));
    }
    return av;
}

// ---------------------------------------------------------------------------
// Kernel 2: rasterize. One wave32 owns a 16-pixel horizontal strip.
//   A (16x32 f16)  = exp2-clamped quadratic weights  (VALU + trans pipe)
//   B (32x16 f16)  = pre-swizzled in LDS, one contiguous 32B load per lane
//   C += A*B via v_wmma_f32_16x16x32_f16             (matrix pipe, co-executes)
// Two accumulators break the WMMA->WMMA RAW chain (no hazard NOPs).
// ---------------------------------------------------------------------------
__global__ __launch_bounds__(256)
void raster_wmma_kernel(const float* __restrict__ gp,
                        float* __restrict__ out, int n) {
    __shared__ float sh[MAXN * 8];                      // 32 KB gaussian table
    __shared__ __align__(32) _Float16 shB[MAXN * 16];   // 32 KB swizzled B

    // Stage full gaussian table into LDS (float4 granularity).
    {
        const float4* g4 = (const float4*)gp;
        float4* s4 = (float4*)sh;
        int total4 = (n * 8) >> 2;
        for (int i = threadIdx.x; i < total4; i += blockDim.x) s4[i] = g4[i];
    }
    __syncthreads();

    // Build B in exact WMMA-B VGPR order: entry for (chunk cb, lane L, slot j)
    // is B[K][N] with N = L&15 (color channel, zero for N>=3) and
    // K = kbase + (j&7) + (j>=8 ? 16 : 0) + (L>=16 ? 8 : 0).  Gaussian g = cb*32 + kk appears at
    // j = (kk&7) | ((kk&16)>>1), lanes L = ((kk&8) ? 16 : 0) + col.
    for (int g = threadIdx.x; g < n; g += blockDim.x) {
        int kk = g & 31;
        int j  = (kk & 7) | ((kk & 16) >> 1);
        int lbase = (kk & 8) ? 16 : 0;
        int cb32  = g & ~31;            // cb*32
        float pm0 = sh[g * 8 + 5];
        float pm1 = sh[g * 8 + 6];
        float pm2 = sh[g * 8 + 7];
#pragma unroll
        for (int col = 0; col < 16; ++col) {
            float v = (col == 0) ? pm0 : (col == 1) ? pm1 : (col == 2) ? pm2 : 0.0f;
            shB[(cb32 + lbase + col) * 16 + j] = (_Float16)v;
        }
    }
    __syncthreads();

    const int lane  = threadIdx.x & 31;
    const int wv    = threadIdx.x >> 5;
    const int strip = blockIdx.x * WAVES_PER_BLOCK + wv;
    if (strip >= TOTAL_STRIPS) return;          // wave-uniform exit

    const int y  = strip / STRIPS_PER_ROW;
    const int xs = (strip % STRIPS_PER_ROW) * 16;

    const int  hioff = (lane >= 16) ? 8 : 0;
    const int  col   = lane & 15;               // B/D column = color channel
    const float px = (float)(xs + col);         // this lane's A-row pixel x
    const float py = (float)y;

    v8f acc0 = {}, acc1 = {};
    const int chunks = n >> 5;                  // n = 1024 -> 32 chunks

    int cb = 0;
    for (; cb + 1 < chunks; cb += 2) {
        int kb0 = (cb + 0) << 5;
        int kb1 = (cb + 1) << 5;
        v16h bv0 = *(const v16h*)&shB[(kb0 + lane) * 16];
        v16h bv1 = *(const v16h*)&shB[(kb1 + lane) * 16];
        v16h av0 = weight_tile(sh, kb0, hioff, px, py);
        acc0 = __builtin_amdgcn_wmma_f32_16x16x32_f16(
            false, av0, false, bv0, (short)0, acc0, false, false);
        v16h av1 = weight_tile(sh, kb1, hioff, px, py);
        acc1 = __builtin_amdgcn_wmma_f32_16x16x32_f16(
            false, av1, false, bv1, (short)0, acc1, false, false);
    }
    if (cb < chunks) {                          // odd-chunk tail
        int kb0 = cb << 5;
        v16h bv0 = *(const v16h*)&shB[(kb0 + lane) * 16];
        v16h av0 = weight_tile(sh, kb0, hioff, px, py);
        acc0 = __builtin_amdgcn_wmma_f32_16x16x32_f16(
            false, av0, false, bv0, (short)0, acc0, false, false);
    }

    // --- Write D: lane holds rows M = r + (lane>=16 ? 8 : 0), column `col` ---
    if (col < 3) {
        const int mbase = (lane >= 16) ? 8 : 0;
#pragma unroll
        for (int r = 0; r < 8; ++r) {
            int x = xs + mbase + r;
            out[col * (IMG_H * IMG_W) + y * IMG_W + x] = acc0[r] + acc1[r];
        }
    }
}

// ---------------------------------------------------------------------------
extern "C" void kernel_launch(void* const* d_in, const int* in_sizes, int n_in,
                              void* d_out, int out_size, void* d_ws, size_t ws_size,
                              hipStream_t stream) {
    const float* alpha    = (const float*)d_in[0];
    const float* color    = (const float*)d_in[1];
    const float* offset   = (const float*)d_in[2];
    const float* scale    = (const float*)d_in[3];
    const float* rotation = (const float*)d_in[4];
    const float* basep    = (const float*)d_in[5];
    const float* imp      = (const float*)d_in[6];
    int n = in_sizes[0];                 // N = 1024
    if (n > MAXN) n = MAXN;

    float* gp  = (float*)d_ws;           // 8 floats per gaussian (32 KB)
    float* out = (float*)d_out;          // [3, 224, 224]

    gauss_prep_kernel<<<(n + 255) / 256, 256, 0, stream>>>(
        alpha, color, offset, scale, rotation, basep, imp, gp, n);

    int nblocks = (TOTAL_STRIPS + WAVES_PER_BLOCK - 1) / WAVES_PER_BLOCK; // 392
    raster_wmma_kernel<<<nblocks, 256, 0, stream>>>(gp, out, n);
}
